// GRU_1975684956951
// MI455X (gfx1250) — matrix operand order repaired and checked by reading, not compiled, not run on hardware
//
#include <hip/hip_runtime.h>
#include <hip/hip_bf16.h>
#include <math.h>

#define SEQ    512
#define BATCH  64
#define IN_DIM 512
#define HID    1024
#define NBLK   32   // persistent workgroups in recurrent kernel

typedef __attribute__((ext_vector_type(16))) __bf16       v16bf;
typedef __attribute__((ext_vector_type(8)))  float        v8f;
typedef __attribute__((ext_vector_type(4)))  unsigned int u32x4;
typedef __attribute__((ext_vector_type(4)))  float        f32x4;
typedef unsigned short u16;
typedef __attribute__((ext_vector_type(4)))  u16          u16x4;

// ---------------- helpers ----------------

__device__ __forceinline__ u16 f2bf(float f) {
  unsigned u = __float_as_uint(f);
  unsigned r = 0x7FFFu + ((u >> 16) & 1u);   // round-to-nearest-even
  return (u16)((u + r) >> 16);
}

union FragU { u32x4 q[2]; v16bf v; };

// A-matrix 16x32 bf16 fragment (MxK). lane<16 -> row=lane, K=0..15;
// lane>=16 -> row=lane-16, K=16..31 (matches load_b). Source is row-major [row][k].
__device__ __forceinline__ v16bf load_a(const u16* __restrict__ base, int ld, int row0, int k0) {
  unsigned lane = threadIdx.x & 31u;
  const u16* p = base + (size_t)(row0 + (int)(lane & 15u)) * ld + k0 + (int)((lane >> 4) << 4);
  FragU f;
  f.q[0] = *(const u32x4*)(p);        // K = kb+0 .. kb+7
  f.q[1] = *(const u32x4*)(p + 8);    // K = kb+8 .. kb+15
  return f.v;
}

// B-matrix 32x16 bf16 fragment (KxN). Lanes 0-15: K=0..15, lanes 16-31: K=16..31,
// column N = lane&15, 16 contiguous K per lane. Source is pre-transposed [n][k].
__device__ __forceinline__ v16bf load_b(const u16* __restrict__ base, int ld, int col0, int k0) {
  unsigned lane = threadIdx.x & 31u;
  const u16* p = base + (size_t)(col0 + (int)(lane & 15u)) * ld + k0 + (int)((lane >> 4) << 4);
  FragU f;
  f.q[0] = *(const u32x4*)(p);        // K = kb+0 .. kb+7
  f.q[1] = *(const u32x4*)(p + 8);    // K = kb+8 .. kb+15
  return f.v;
}

__device__ __forceinline__ v8f wmma_bf16(v16bf a, v16bf b, v8f c) {
  return __builtin_amdgcn_wmma_f32_16x16x32_bf16(false, a, false, b, (short)0, c, false, false);
}

__device__ __forceinline__ float sigmoidf_fast(float x) {
  return 1.0f / (1.0f + __expf(-x));
}

// Device-wide generation barrier. fences lower to global_wb / global_inv on gfx1250
// so cross-WGP data written before the barrier is visible after it.
__device__ __forceinline__ void gridBarrier(unsigned* cnt, unsigned* gen, unsigned nb) {
  __builtin_amdgcn_fence(__ATOMIC_RELEASE, "agent");
  __syncthreads();
  if (threadIdx.x == 0) {
    unsigned g = __hip_atomic_load(gen, __ATOMIC_RELAXED, __HIP_MEMORY_SCOPE_AGENT);
    unsigned arrived = __hip_atomic_fetch_add(cnt, 1u, __ATOMIC_RELAXED, __HIP_MEMORY_SCOPE_AGENT);
    if (arrived == nb - 1u) {
      __hip_atomic_store(cnt, 0u, __ATOMIC_RELAXED, __HIP_MEMORY_SCOPE_AGENT);
      __hip_atomic_fetch_add(gen, 1u, __ATOMIC_RELEASE, __HIP_MEMORY_SCOPE_AGENT);
    } else {
      while (__hip_atomic_load(gen, __ATOMIC_RELAXED, __HIP_MEMORY_SCOPE_AGENT) == g) {
        __builtin_amdgcn_s_sleep(2);
      }
    }
  }
  __syncthreads();
  __builtin_amdgcn_fence(__ATOMIC_ACQUIRE, "agent");
}

// ---------------- prep kernels ----------------

// f32 embedding -> bf16 (flat [SEQ*BATCH][IN_DIM]), 4-wide
__global__ void k_cvt_emb(const float* __restrict__ src, u16* __restrict__ dst, int n4) {
  int i = blockIdx.x * blockDim.x + threadIdx.x;
  if (i >= n4) return;
  f32x4 v = *(const f32x4*)(src + (size_t)i * 4);
  u16x4 o;
  o.x = f2bf(v.x); o.y = f2bf(v.y); o.z = f2bf(v.z); o.w = f2bf(v.w);
  *(u16x4*)(dst + (size_t)i * 4) = o;
}

// split + transpose + convert weights: W[k][n] (f32, [IN+HID][HID]) ->
// WxT[w][n][k] (k<IN) and WhT[w][n][k-IN] (k>=IN), bf16
__global__ void k_cvt_w(const float* __restrict__ Wz, const float* __restrict__ Wr,
                        const float* __restrict__ Wm,
                        u16* __restrict__ WxT, u16* __restrict__ WhT) {
  const int per = (IN_DIM + HID) * HID;
  int idx = blockIdx.x * blockDim.x + threadIdx.x;
  if (idx >= 3 * per) return;
  int w = idx / per;
  int r = idx - w * per;
  int k = r / HID;
  int n = r - k * HID;
  const float* W = (w == 0) ? Wz : (w == 1) ? Wr : Wm;
  u16 v = f2bf(W[(size_t)k * HID + n]);
  if (k < IN_DIM)
    WxT[(size_t)w * HID * IN_DIM + (size_t)n * IN_DIM + k] = v;
  else
    WhT[(size_t)w * HID * HID + (size_t)n * HID + (k - IN_DIM)] = v;
}

// init_h -> h_f32 + h_bf16; zero barrier state
__global__ void k_init(const float* __restrict__ init_h, float* __restrict__ h_f32,
                       u16* __restrict__ h_bf16, unsigned* __restrict__ bar, int n) {
  int i = blockIdx.x * blockDim.x + threadIdx.x;
  if (i == 0) { bar[0] = 0u; bar[1] = 0u; }
  if (i < n) {
    float v = init_h[i];
    h_f32[i]  = v;
    h_bf16[i] = f2bf(v);
  }
}

// ---------------- input projection GEMM ----------------
// X[w][s][b][h] = emb[s][b][:] @ Wx[w][:][h] + bias[w][h]
// block = 256 threads = 8 waves, tile 64(M) x 32(N); grid (M/64, N/32, 3)
__global__ __launch_bounds__(256) void k_xproj(
    const u16* __restrict__ embB, const u16* __restrict__ WxT,
    const float* __restrict__ bz, const float* __restrict__ br,
    const float* __restrict__ bm, float* __restrict__ X) {
  int w = blockIdx.z;
  const u16*   Wt   = WxT + (size_t)w * HID * IN_DIM;
  const float* bias = (w == 0) ? bz : (w == 1) ? br : bm;
  float*       Xw   = X + (size_t)w * SEQ * BATCH * HID;

  int wave = threadIdx.x >> 5;
  int lane = threadIdx.x & 31;
  int m0 = blockIdx.x * 64 + (wave >> 1) * 16;
  int n0 = blockIdx.y * 32 + (wave & 1) * 16;

  v8f acc = {};
#pragma unroll 4
  for (int k = 0; k < IN_DIM; k += 32) {
    v16bf a = load_a(embB, IN_DIM, m0, k);
    v16bf b = load_b(Wt,   IN_DIM, n0, k);
    acc = wmma_bf16(a, b, acc);
  }
  int n     = n0 + (lane & 15);
  int mbase = m0 + ((lane >> 4) << 3);
  float bv = bias[n];
#pragma unroll
  for (int j = 0; j < 8; ++j)
    Xw[(size_t)(mbase + j) * HID + n] = acc[j] + bv;
}

// ---------------- persistent recurrent kernel ----------------
// 32 workgroups (one per 32 H-columns), 8 waves each; wave tile 16x16.
// Per step: phase1 z,r gates -> rh; grid barrier; phase2 m-GEMM on rh, h update; barrier.
__global__ __launch_bounds__(256) void k_gru(
    const u16* __restrict__ WhT,      // [3][HID n][HID k] bf16
    const float* __restrict__ X,      // [3][SEQ][BATCH][HID] f32
    float* __restrict__ h_f32, u16* __restrict__ h_bf16,
    u16* __restrict__ rh_bf16, float* __restrict__ out,
    unsigned* __restrict__ bar) {
  const u16* WzT = WhT;
  const u16* WrT = WhT + (size_t)HID * HID;
  const u16* WmT = WhT + (size_t)2 * HID * HID;
  const float* Xz = X;
  const float* Xr = X + (size_t)SEQ * BATCH * HID;
  const float* Xm = X + (size_t)2 * SEQ * BATCH * HID;

  int wave = threadIdx.x >> 5;
  int lane = threadIdx.x & 31;
  int m0 = (wave >> 1) * 16;                  // batch-row tile (BATCH=64 -> 4 tiles)
  int n0 = blockIdx.x * 32 + (wave & 1) * 16; // H-column tile
  int n     = n0 + (lane & 15);
  int mbase = m0 + ((lane >> 4) << 3);

  for (int s = 0; s < SEQ; ++s) {
    const float* Xzs = Xz + (size_t)s * BATCH * HID;
    const float* Xrs = Xr + (size_t)s * BATCH * HID;
    const float* Xms = Xm + (size_t)s * BATCH * HID;
    __builtin_prefetch(Xms + (size_t)mbase * HID + n, 0, 1); // global_prefetch_b8

    // ---- phase 1: z = sig(xz + h@Wz_h), r = sig(xr + h@Wr_h), rh = r*h ----
    v8f az = {}, ar = {};
    for (int k = 0; k < HID; k += 32) {
      v16bf a  = load_a(h_bf16, HID, m0, k);
      v16bf bz = load_b(WzT,    HID, n0, k);
      az = wmma_bf16(a, bz, az);
      v16bf br_ = load_b(WrT,   HID, n0, k);
      ar = wmma_bf16(a, br_, ar);
    }
    float zreg[8], hreg[8];
#pragma unroll
    for (int j = 0; j < 8; ++j) {
      size_t off = (size_t)(mbase + j) * HID + n;
      float hv = h_f32[off];
      float zf = sigmoidf_fast(Xzs[off] + az[j]);
      float rf = sigmoidf_fast(Xrs[off] + ar[j]);
      rh_bf16[off] = f2bf(rf * hv);
      zreg[j] = zf; hreg[j] = hv;
    }
    gridBarrier(&bar[0], &bar[1], NBLK);

    // ---- phase 2: h_tilde = tanh(xm + (r*h)@Wm_h); h = (1-z)h + z h_tilde ----
    v8f am = {};
    for (int k = 0; k < HID; k += 32) {
      v16bf a  = load_a(rh_bf16, HID, m0, k);
      v16bf bm = load_b(WmT,     HID, n0, k);
      am = wmma_bf16(a, bm, am);
    }
#pragma unroll
    for (int j = 0; j < 8; ++j) {
      int m = mbase + j;
      size_t off = (size_t)m * HID + n;
      float ht = tanhf(Xms[off] + am[j]);
      float hn = (1.0f - zreg[j]) * hreg[j] + zreg[j] * ht;
      h_f32[off]  = hn;
      h_bf16[off] = f2bf(hn);
      out[(size_t)m * SEQ * HID + (size_t)s * HID + n] = hn;
    }
    gridBarrier(&bar[0], &bar[1], NBLK);
  }
}

// ---------------- host launcher ----------------

extern "C" void kernel_launch(void* const* d_in, const int* in_sizes, int n_in,
                              void* d_out, int out_size, void* d_ws, size_t ws_size,
                              hipStream_t stream) {
  (void)in_sizes; (void)n_in; (void)out_size; (void)ws_size;
  const float* emb    = (const float*)d_in[0];
  const float* init_h = (const float*)d_in[1];
  const float* Wz     = (const float*)d_in[2];
  const float* bz     = (const float*)d_in[3];
  const float* Wr     = (const float*)d_in[4];
  const float* br     = (const float*)d_in[5];
  const float* Wm     = (const float*)d_in[6];
  const float* bm     = (const float*)d_in[7];
  float* out = (float*)d_out;

  char* ws = (char*)d_ws;
  size_t off = 0;
  auto alloc = [&](size_t bytes) -> void* {
    void* p = ws + off;
    off = (off + bytes + 255) & ~(size_t)255;
    return p;
  };
  u16*      embB   = (u16*)alloc((size_t)SEQ * BATCH * IN_DIM * 2);   // 32 MB
  u16*      WxT    = (u16*)alloc((size_t)3 * HID * IN_DIM * 2);       // 3 MB
  u16*      WhT    = (u16*)alloc((size_t)3 * HID * HID * 2);          // 6 MB
  float*    X      = (float*)alloc((size_t)3 * SEQ * BATCH * HID * 4);// 384 MB
  float*    h_f32  = (float*)alloc((size_t)BATCH * HID * 4);
  u16*      h_bf16 = (u16*)alloc((size_t)BATCH * HID * 2);
  u16*      rh     = (u16*)alloc((size_t)BATCH * HID * 2);
  unsigned* bar    = (unsigned*)alloc(256);

  // prep
  {
    int n4 = SEQ * BATCH * IN_DIM / 4;
    k_cvt_emb<<<(n4 + 255) / 256, 256, 0, stream>>>(emb, embB, n4);
  }
  {
    int tot = 3 * (IN_DIM + HID) * HID;
    k_cvt_w<<<(tot + 255) / 256, 256, 0, stream>>>(Wz, Wr, Wm, WxT, WhT);
  }
  {
    int n = BATCH * HID;
    k_init<<<(n + 255) / 256, 256, 0, stream>>>(init_h, h_f32, h_bf16, bar, n);
  }
  // input projections (big parallel WMMA GEMM)
  {
    dim3 grid(SEQ * BATCH / 64, HID / 32, 3);
    k_xproj<<<grid, 256, 0, stream>>>(embB, WxT, bz, br, bm, X);
  }
  // serial recurrence (persistent, grid-synchronized)
  k_gru<<<NBLK, 256, 0, stream>>>(WhT, X, h_f32, h_bf16, rh, out, bar);
}
